// BaseAttention_19842748907813
// MI455X (gfx1250) — matrix operand order repaired and checked by reading, not compiled, not run on hardware
//
#include <hip/hip_runtime.h>
#include <hip/hip_bf16.h>

typedef __bf16 bf16;
typedef __attribute__((ext_vector_type(8)))  __bf16 v8bf;
typedef __attribute__((ext_vector_type(16))) __bf16 v16bf;
typedef __attribute__((ext_vector_type(8)))  float  v8f;

#define B_    2
#define S_    2048
#define D_    2048
#define H_    16
#define KVH_  4
#define HD_   128
#define R_    128
#define WIN_  512
#define GLB_  64
#define MROWS (B_ * S_)   // 4096

// ---------------------------------------------------------------- helpers

static __device__ inline v8bf ldg8(const bf16* p) {
    return *reinterpret_cast<const v8bf*>(p);
}
static __device__ inline v16bf cat16(v8bf a, v8bf b) {
    return __builtin_shufflevector(a, b, 0,1,2,3,4,5,6,7,8,9,10,11,12,13,14,15);
}
static __device__ inline v8f wmma_bf16(v16bf a, v16bf b, v8f c) {
    // D = A(16x32) * B(32x16) + C ; emits v_wmma_f32_16x16x32_bf16
    return __builtin_amdgcn_wmma_f32_16x16x32_bf16(
        /*neg_a=*/false, a, /*neg_b=*/false, b,
        /*c_mod=*/(short)0, c, /*reuse_a=*/false, /*reuse_b=*/false);
}

// A tile: rows m0..m0+15 of row-major X (ld), K-chunk k0..k0+31.
// A layout (matches B): lanes 0-15 hold K 0-15; lanes 16-31 hold K 16-31.
static __device__ inline v16bf load_a(const bf16* X, int ld, int m0, int k0, int lane) {
    int r = lane & 15, half = lane >> 4;
    const bf16* p = X + (size_t)(m0 + r) * ld + k0 + half * 16;
    return cat16(ldg8(p), ldg8(p + 8));
}
// B tile: column n of B = row (n0+n) of row-major W (ld), K-chunk k0..k0+31.
// B layout (contiguous, matches A): lanes 0-15 hold K 0-15; lanes 16-31 hold K 16-31.
static __device__ inline v16bf load_b(const bf16* W, int ld, int n0, int k0, int lane) {
    int n = lane & 15, half = lane >> 4;
    const bf16* p = W + (size_t)(n0 + n) * ld + k0 + half * 16;
    return cat16(ldg8(p), ldg8(p + 8));
}

// ---------------------------------------------------------------- kernels

// Weff[n,k] = W[n,k] + sum_r Bm[n,r] * A[r,k]   (fold LoRA into dense weight)
__global__ void prep_weight(const float* __restrict__ W, const float* __restrict__ A,
                            const float* __restrict__ Bm, bf16* __restrict__ Weff,
                            int N, int K) {
    int idx = blockIdx.x * blockDim.x + threadIdx.x;
    if (idx >= N * K) return;
    int n = idx / K, d = idx - n * K;
    float acc = W[idx];
    const float* brow = Bm + (size_t)n * R_;
#pragma unroll 8
    for (int r = 0; r < R_; ++r) acc += brow[r] * A[(size_t)r * K + d];
    Weff[idx] = (bf16)acc;
}

__global__ void cast_f2b(const float* __restrict__ in, bf16* __restrict__ out, int n) {
    int i = blockIdx.x * blockDim.x + threadIdx.x;
    if (i < n) out[i] = (bf16)in[i];
}

// Y(MxN fp32) = X(MxK bf16) * W(NxK bf16)^T.
// Block tile: 128(M) x 64(N); 4 waves, each a 32x64 tile (2 M-subtiles).
// B panel (64 rows x 32 K, 4KB) shared by all 4 waves, staged into a
// double-buffered LDS panel with global_load_async_to_lds_b128 (ASYNCcnt),
// overlapped with WMMA on the previous panel. A operands software-pipelined
// one K-chunk ahead in registers. Last chunk peeled so the steady-state loop
// stages unconditionally (no register-rotation branches).
__global__ void __launch_bounds__(128)
gemm_xwt(const bf16* __restrict__ X, const bf16* __restrict__ W,
         float* __restrict__ Y, int M, int N, int K) {
    __shared__ __align__(16) bf16 Bs[2][64][32];   // 8 KB double buffer
    int lane = threadIdx.x & 31;
    int wave = threadIdx.x >> 5;
    int m0 = blockIdx.x * 128 + wave * 32;
    int n0 = blockIdx.y * 64;
    int r = lane & 15, half = lane >> 4;

    // byte offset of Bs within LDS (flat LDS pointer low 32 bits == LDS offset)
    uint32_t ldsBase = (uint32_t)(uintptr_t)(&Bs[0][0][0]);
    int sn = threadIdx.x >> 1;            // staging row 0..63
    int skb = (threadIdx.x & 1) * 16;     // staging K element offset 0/16

    // issue async global->LDS copy of one 64x32 B panel (2 instrs/wave)
    auto stage = [&](int buf, int k0) {
        const bf16* g = W + (size_t)(n0 + sn) * K + k0 + skb;
        uint32_t l = ldsBase + (uint32_t)buf * 4096u + (uint32_t)((sn * 32 + skb) * 2);
        asm volatile("global_load_async_to_lds_b128 %0, %1, off"
                     :: "v"(l), "v"(g) : "memory");
        asm volatile("global_load_async_to_lds_b128 %0, %1, off offset:16"
                     :: "v"(l), "v"(g) : "memory");
    };
    // read one 32x16 B fragment (n-tile t) from LDS panel buf
    auto ldsB = [&](int buf, int t) -> v16bf {
        const bf16* p = &Bs[buf][t * 16 + r][half * 16];
        return cat16(*reinterpret_cast<const v8bf*>(p),
                     *reinterpret_cast<const v8bf*>(p + 8));
    };

    v8f acc0[4], acc1[4];
#pragma unroll
    for (int t = 0; t < 4; ++t) { acc0[t] = (v8f){}; acc1[t] = (v8f){}; }

    int nch = K / 32;                      // >= 2 for all our GEMMs
    stage(0, 0);
    v16bf a0 = load_a(X, K, m0, 0, lane);
    v16bf a1 = load_a(X, K, m0 + 16, 0, lane);

    // steady state: always stage the next panel, consume the current one
#pragma unroll 2
    for (int c = 0; c < nch - 1; ++c) {
        int buf = c & 1;
        int k0 = c * 32;
        stage(buf ^ 1, k0 + 32);                        // async fill of next panel
        v16bf na0 = load_a(X, K, m0, k0 + 32, lane);    // next A in flight
        v16bf na1 = load_a(X, K, m0 + 16, k0 + 32, lane);
        asm volatile("s_wait_asynccnt 0x2" ::: "memory");   // current panel done
        __syncthreads();                                // panel visible to all waves
        v16bf b0 = ldsB(buf, 0);
        v16bf b1 = ldsB(buf, 1);
        v16bf b2 = ldsB(buf, 2);
        v16bf b3 = ldsB(buf, 3);
        acc0[0] = wmma_bf16(a0, b0, acc0[0]);
        acc1[0] = wmma_bf16(a1, b0, acc1[0]);
        acc0[1] = wmma_bf16(a0, b1, acc0[1]);
        acc1[1] = wmma_bf16(a1, b1, acc1[1]);
        acc0[2] = wmma_bf16(a0, b2, acc0[2]);
        acc1[2] = wmma_bf16(a1, b2, acc1[2]);
        acc0[3] = wmma_bf16(a0, b3, acc0[3]);
        acc1[3] = wmma_bf16(a1, b3, acc1[3]);
        __syncthreads();                                // all waves done reading panel
        a0 = na0; a1 = na1;
    }
    // epilogue: last chunk (already staged by final loop iteration or prologue)
    {
        int buf = (nch - 1) & 1;
        asm volatile("s_wait_asynccnt 0x0" ::: "memory");
        __syncthreads();
        v16bf b0 = ldsB(buf, 0);
        v16bf b1 = ldsB(buf, 1);
        v16bf b2 = ldsB(buf, 2);
        v16bf b3 = ldsB(buf, 3);
        acc0[0] = wmma_bf16(a0, b0, acc0[0]);
        acc1[0] = wmma_bf16(a1, b0, acc1[0]);
        acc0[1] = wmma_bf16(a0, b1, acc0[1]);
        acc1[1] = wmma_bf16(a1, b1, acc1[1]);
        acc0[2] = wmma_bf16(a0, b2, acc0[2]);
        acc1[2] = wmma_bf16(a1, b2, acc1[2]);
        acc0[3] = wmma_bf16(a0, b3, acc0[3]);
        acc1[3] = wmma_bf16(a1, b3, acc1[3]);
    }

#pragma unroll
    for (int t = 0; t < 4; ++t)
#pragma unroll
        for (int v = 0; v < 8; ++v) {
            int mA = m0 + v + 8 * half;
            int mB = m0 + 16 + v + 8 * half;
            Y[(size_t)mA * N + n0 + t * 16 + r] = acc0[t][v];
            Y[(size_t)mB * N + n0 + t * 16 + r] = acc1[t][v];
        }
}

// RoPE + cast + relayout: in [B][S][heads][HD] fp32 -> out [B][heads][S][HD] bf16
__global__ void rope_pack(const float* __restrict__ in, bf16* __restrict__ out, int heads) {
    int idx = blockIdx.x * blockDim.x + threadIdx.x;
    int total = B_ * S_ * heads * (HD_ / 2);
    if (idx >= total) return;
    int p = idx & 63; int t = idx >> 6;
    int h = t % heads; t /= heads;
    int s = t % S_;    int b = t / S_;
    float inv = __powf(1.0e6f, -(float)p * (1.0f / 64.0f));
    float ang = (float)s * inv;
    float c = cosf(ang), sn = sinf(ang);
    size_t ibase = (((size_t)(b * S_ + s) * heads) + h) * HD_;
    float xr = in[ibase + 2 * p], xi = in[ibase + 2 * p + 1];
    size_t obase = ((((size_t)b * heads + h) * S_) + s) * HD_;
    out[obase + 2 * p]     = (bf16)(xr * c - xi * sn);
    out[obase + 2 * p + 1] = (bf16)(xr * sn + xi * c);
}

// V relayout: in [B][S][KVH][HD] fp32 -> out [B][KVH][HD][S] bf16 (transposed)
__global__ void pack_vt(const float* __restrict__ in, bf16* __restrict__ out) {
    int idx = blockIdx.x * blockDim.x + threadIdx.x;
    if (idx >= B_ * S_ * KVH_ * HD_) return;
    int hd = idx & (HD_ - 1);
    int t  = idx >> 7;
    int kv = t & (KVH_ - 1); t >>= 2;
    int s  = t & (S_ - 1);
    int b  = t >> 11;
    out[((((size_t)b * KVH_ + kv) * HD_) + hd) * S_ + s] = (bf16)in[idx];
}

// Flash attention. Block = (b, kvh, 16-query tile); 4 waves = 4 GQA heads
// sharing identical K/V. Q [B][H][S][HD], K [B][KVH][S][HD], Vt [B][KVH][HD][S].
// Out: bf16 [B*S][H*HD].
__global__ void __launch_bounds__(128)
attn_kernel(const bf16* __restrict__ Q, const bf16* __restrict__ Kt,
            const bf16* __restrict__ Vt, bf16* __restrict__ O) {
    __shared__ __align__(16) bf16 plds[4][16][32];
    int lane = threadIdx.x & 31;
    int wave = threadIdx.x >> 5;
    int s0 = blockIdx.x * 16;
    int kv = blockIdx.y;
    int b  = blockIdx.z;
    int h  = kv * 4 + wave;
    int r = lane & 15, half = lane >> 4;

    const bf16* qbase = Q  + (((size_t)b * H_   + h)  * S_)  * HD_;
    const bf16* kbase = Kt + (((size_t)b * KVH_ + kv) * S_)  * HD_;
    const bf16* vbase = Vt + (((size_t)b * KVH_ + kv) * HD_) * S_;

    v16bf qa[4];
#pragma unroll
    for (int c = 0; c < 4; ++c) {
        const bf16* p = qbase + (size_t)(s0 + r) * HD_ + c * 32 + half * 16;
        qa[c] = cat16(ldg8(p), ldg8(p + 8));
    }

    v8f acc[8];
#pragma unroll
    for (int t = 0; t < 8; ++t) acc[t] = (v8f){};
    v8f m_run, l_run;
#pragma unroll
    for (int v = 0; v < 8; ++v) { m_run[v] = -3.0e38f; l_run[v] = 0.0f; }

    const float scale = 0.08838834764831845f;  // 1/sqrt(128)

    for (int j0 = 0; j0 <= s0 + 15; j0 += 32) {
        // skip key blocks fully outside (global OR sliding-window) region
        if (!((j0 < GLB_) || (j0 + 31 >= s0 - (WIN_ - 1)))) continue;

        // ---- scores: S(16x32) = Q(16x128) * K^T, two 16x16 C tiles
        v8f st0 = {}, st1 = {};
#pragma unroll
        for (int c = 0; c < 4; ++c) {
            const bf16* p0 = kbase + (size_t)(j0 + r) * HD_ + c * 32 + half * 16;
            st0 = wmma_bf16(qa[c], cat16(ldg8(p0), ldg8(p0 + 8)), st0);
            const bf16* p1 = kbase + (size_t)(j0 + 16 + r) * HD_ + c * 32 + half * 16;
            st1 = wmma_bf16(qa[c], cat16(ldg8(p1), ldg8(p1 + 8)), st1);
        }

        // ---- scale + mask, per-element (C layout: row = v + 8*half, col = r)
        v8f bm;
#pragma unroll
        for (int v = 0; v < 8; ++v) {
            int i  = s0 + v + 8 * half;
            int ja = j0 + r;
            int jb = j0 + 16 + r;
            bool ok0 = (ja <= i) && ((i - ja < WIN_) || (ja < GLB_));
            bool ok1 = (jb <= i) && ((i - jb < WIN_) || (jb < GLB_));
            st0[v] = ok0 ? st0[v] * scale : -1.0e9f;
            st1[v] = ok1 ? st1[v] * scale : -1.0e9f;
            bm[v] = fmaxf(st0[v], st1[v]);
        }
        // ---- row max across the 16 lanes of each half-wave
#pragma unroll
        for (int off = 1; off < 16; off <<= 1)
#pragma unroll
            for (int v = 0; v < 8; ++v)
                bm[v] = fmaxf(bm[v], __shfl_xor(bm[v], off, 32));

        v8f sc;
#pragma unroll
        for (int v = 0; v < 8; ++v) {
            float mnew = fmaxf(m_run[v], bm[v]);
            sc[v] = __expf(m_run[v] - mnew);
            st0[v] = __expf(st0[v] - mnew);
            st1[v] = __expf(st1[v] - mnew);
            m_run[v] = mnew;
        }
        // ---- row sum of P
        v8f rs;
#pragma unroll
        for (int v = 0; v < 8; ++v) rs[v] = st0[v] + st1[v];
#pragma unroll
        for (int off = 1; off < 16; off <<= 1)
#pragma unroll
            for (int v = 0; v < 8; ++v) rs[v] += __shfl_xor(rs[v], off, 32);
#pragma unroll
        for (int v = 0; v < 8; ++v) l_run[v] = l_run[v] * sc[v] + rs[v];
        // ---- rescale running O
#pragma unroll
        for (int t = 0; t < 8; ++t)
#pragma unroll
            for (int v = 0; v < 8; ++v) acc[t][v] *= sc[v];

        // ---- P: C layout -> A layout through per-wave LDS (DS is in-order per wave)
#pragma unroll
        for (int v = 0; v < 8; ++v) {
            plds[wave][v + 8 * half][r]      = (bf16)st0[v];
            plds[wave][v + 8 * half][16 + r] = (bf16)st1[v];
        }
        v16bf pa;
        {
            const bf16* p0 = &plds[wave][r][half * 16];
            const bf16* p1 = &plds[wave][r][half * 16 + 8];
            pa = cat16(*reinterpret_cast<const v8bf*>(p0),
                       *reinterpret_cast<const v8bf*>(p1));
        }
        // ---- O += P(16x32) * V(32x128): Vt rows are contiguous in keys
#pragma unroll
        for (int t = 0; t < 8; ++t) {
            const bf16* p = vbase + (size_t)(t * 16 + r) * S_ + j0 + half * 16;
            acc[t] = wmma_bf16(pa, cat16(ldg8(p), ldg8(p + 8)), acc[t]);
        }
    }

    // ---- normalize + store bf16 into [B*S][H*HD]
#pragma unroll
    for (int t = 0; t < 8; ++t)
#pragma unroll
        for (int v = 0; v < 8; ++v) {
            int m = v + 8 * half;
            size_t orow = ((size_t)(b * S_ + s0 + m)) * (H_ * HD_) + (size_t)h * HD_;
            O[orow + t * 16 + r] = (bf16)(acc[t][v] / l_run[v]);
        }
}

// ---------------------------------------------------------------- launch

extern "C" void kernel_launch(void* const* d_in, const int* in_sizes, int n_in,
                              void* d_out, int out_size, void* d_ws, size_t ws_size,
                              hipStream_t stream) {
    const float* x    = (const float*)d_in[0];
    const float* wq_w = (const float*)d_in[1];
    const float* wq_a = (const float*)d_in[2];
    const float* wq_b = (const float*)d_in[3];
    const float* wk_w = (const float*)d_in[4];
    const float* wk_a = (const float*)d_in[5];
    const float* wk_b = (const float*)d_in[6];
    const float* wv_w = (const float*)d_in[7];
    const float* wv_a = (const float*)d_in[8];
    const float* wv_b = (const float*)d_in[9];
    const float* wo_w = (const float*)d_in[10];
    const float* wo_a = (const float*)d_in[11];
    const float* wo_b = (const float*)d_in[12];

    char* ws = (char*)d_ws;
    size_t off = 0;
    auto alloc = [&](size_t bytes) -> void* {
        void* p = ws + off;
        off += (bytes + 255) & ~(size_t)255;
        return p;
    };

    bf16* xb     = (bf16*)alloc((size_t)MROWS * D_ * 2);
    bf16* wq_eff = (bf16*)alloc((size_t)(H_ * HD_) * D_ * 2);
    bf16* wk_eff = (bf16*)alloc((size_t)(KVH_ * HD_) * D_ * 2);
    bf16* wv_eff = (bf16*)alloc((size_t)(KVH_ * HD_) * D_ * 2);
    bf16* wo_eff = (bf16*)alloc((size_t)D_ * (H_ * HD_) * 2);
    float* qf    = (float*)alloc((size_t)MROWS * (H_ * HD_) * 4);
    float* kf    = (float*)alloc((size_t)MROWS * (KVH_ * HD_) * 4);
    float* vf    = (float*)alloc((size_t)MROWS * (KVH_ * HD_) * 4);
    bf16* qbf    = (bf16*)alloc((size_t)MROWS * (H_ * HD_) * 2);
    bf16* kbf    = (bf16*)alloc((size_t)MROWS * (KVH_ * HD_) * 2);
    bf16* vtbf   = (bf16*)alloc((size_t)MROWS * (KVH_ * HD_) * 2);
    bf16* attnb  = (bf16*)alloc((size_t)MROWS * (H_ * HD_) * 2);

    // 1) fold LoRA into dense weights (bf16)
    {
        int nq = (H_ * HD_) * D_;
        int nk = (KVH_ * HD_) * D_;
        prep_weight<<<(nq + 255) / 256, 256, 0, stream>>>(wq_w, wq_a, wq_b, wq_eff, H_ * HD_, D_);
        prep_weight<<<(nk + 255) / 256, 256, 0, stream>>>(wk_w, wk_a, wk_b, wk_eff, KVH_ * HD_, D_);
        prep_weight<<<(nk + 255) / 256, 256, 0, stream>>>(wv_w, wv_a, wv_b, wv_eff, KVH_ * HD_, D_);
        prep_weight<<<(nq + 255) / 256, 256, 0, stream>>>(wo_w, wo_a, wo_b, wo_eff, D_, H_ * HD_);
    }
    // 2) x -> bf16
    {
        int n = MROWS * D_;
        cast_f2b<<<(n + 255) / 256, 256, 0, stream>>>(x, xb, n);
    }
    // 3) Q/K/V projections (WMMA GEMM, async-LDS B panels)
    gemm_xwt<<<dim3(MROWS / 128, (H_ * HD_) / 64), 128, 0, stream>>>(xb, wq_eff, qf, MROWS, H_ * HD_, D_);
    gemm_xwt<<<dim3(MROWS / 128, (KVH_ * HD_) / 64), 128, 0, stream>>>(xb, wk_eff, kf, MROWS, KVH_ * HD_, D_);
    gemm_xwt<<<dim3(MROWS / 128, (KVH_ * HD_) / 64), 128, 0, stream>>>(xb, wv_eff, vf, MROWS, KVH_ * HD_, D_);
    // 4) RoPE + relayout
    {
        int nq = B_ * S_ * H_ * (HD_ / 2);
        int nk = B_ * S_ * KVH_ * (HD_ / 2);
        int nv = B_ * S_ * KVH_ * HD_;
        rope_pack<<<(nq + 255) / 256, 256, 0, stream>>>(qf, qbf, H_);
        rope_pack<<<(nk + 255) / 256, 256, 0, stream>>>(kf, kbf, KVH_);
        pack_vt<<<(nv + 255) / 256, 256, 0, stream>>>(vf, vtbf);
    }
    // 5) flash attention (WMMA)
    attn_kernel<<<dim3(S_ / 16, KVH_, B_), 128, 0, stream>>>(qbf, kbf, vtbf, attnb);
    // 6) output projection -> d_out (fp32)
    gemm_xwt<<<dim3(MROWS / 128, D_ / 64), 128, 0, stream>>>(attnb, wo_eff, (float*)d_out, MROWS, D_, H_ * HD_);

    (void)in_sizes; (void)n_in; (void)out_size; (void)ws_size;
}
